// P4Transformer_78786880078206
// MI455X (gfx1250) — compile-verified, matrix operand order checked by reading, not run on hardware
//
#include <hip/hip_runtime.h>
#include <hip/hip_bf16.h>
#include <math.h>

// ---------------- problem constants ----------------
#define BV      2
#define TT      8
#define NPT     2048
#define MA      256          // anchors per frame
#define TQ      4
#define KNB     16           // neighbors
#define DIMM    1024
#define SREAL   1041         // 17 + 4*256
#define SPAD    1056         // 66*16, also 33*32
#define JNJ     17
#define NHEADS  8
#define DHEADN  128
#define MLPN    2048
#define EPSV    1e-5f
#define ATT_SCALE 0.08838834764831845f   // 128^-0.5
#define RAD2    0.49f

typedef _Float16 v8h  __attribute__((ext_vector_type(8)));
typedef _Float16 v16h __attribute__((ext_vector_type(16)));
typedef float    v8f  __attribute__((ext_vector_type(8)));

// ---------------- block reduce helpers ----------------
__device__ __forceinline__ float block_sum(float v, float* sm) {
  int tid = threadIdx.x;
  sm[tid] = v; __syncthreads();
  for (int s = blockDim.x >> 1; s > 0; s >>= 1) {
    if (tid < s) sm[tid] += sm[tid + s];
    __syncthreads();
  }
  float r = sm[0]; __syncthreads();
  return r;
}
__device__ __forceinline__ float block_max(float v, float* sm) {
  int tid = threadIdx.x;
  sm[tid] = v; __syncthreads();
  for (int s = blockDim.x >> 1; s > 0; s >>= 1) {
    if (tid < s) sm[tid] = fmaxf(sm[tid], sm[tid + s]);
    __syncthreads();
  }
  float r = sm[0]; __syncthreads();
  return r;
}

// ---------------- WMMA fragment load ----------------
// CDNA5 16-bit A-fragment (16x32): lane l<16 -> row M=l, K=k0+[0..7] (v0..3) and
// k0+[16..23] (v4..7); lane l>=16 -> row M=l-16, K=k0+[8..15] and k0+[24..31].
// Two contiguous 16B loads per lane. B fragment uses the same pattern on the
// N-major (transposed) operand.
__device__ __forceinline__ v16h frag_ld(const _Float16* p, int ld, int r0, int k0, int lane) {
  int l  = lane & 15;
  int hi = lane >> 4;
  const _Float16* q = p + (size_t)(r0 + l) * (size_t)ld + k0 + hi * 8;
  union { v16h v; v8h h[2]; } u;
  u.h[0] = *(const v8h*)q;
  u.h[1] = *(const v8h*)(q + 16);
  return u.v;
}

// ---- K-loop body specialized on live N-subtile count: no conditional
// accumulator updates -> no v_mov accumulator shuffling around the WMMAs. ----
template <int NT>
__device__ __forceinline__ void
gemm_kloop(const _Float16* __restrict__ Ap, const _Float16* __restrict__ Bp,
           v8f (&acc)[4], int m0, int n0, int Kk, int lda, int ldb, int lane)
{
  for (int k0 = 0; k0 < Kk; k0 += 32) {
    v16h a = frag_ld(Ap, lda, m0, k0, lane);
    if (k0 + 32 < Kk)
      __builtin_prefetch(Ap + (size_t)(m0 + (lane & 15)) * (size_t)lda + k0 + 32, 0, 1);
#pragma unroll
    for (int t = 0; t < NT; ++t) {
      v16h bf = frag_ld(Bp, ldb, n0 + t * 16, k0, lane);
      acc[t] = __builtin_amdgcn_wmma_f32_16x16x32_f16(
          false, a, false, bf, (short)0, acc[t], false, false);
    }
  }
}

// ---------------- generic WMMA GEMM ----------------
// C[z] = act(A[z] @ Bt[z]^T + bias) + Res ; A: MxK row-major f16 (lda),
// Bt: NxK row-major f16 (ldb), out f32 or f16 (flags bit0), gelu (bit1).
// Batch z = b*hdiv + h with per-b / per-h element strides.
__global__ void __launch_bounds__(128)
gemm_wmma_kernel(const _Float16* __restrict__ A, const _Float16* __restrict__ Bt,
                 const float* __restrict__ bias, const float* __restrict__ Res,
                 void* __restrict__ C,
                 int Mm, int Nn, int Kk, int lda, int ldb, int ldc, int hdiv,
                 long sAb, long sAh, long sBb, long sBh, long sCb, long sCh,
                 int flags)
{
  int lane = threadIdx.x & 31;
  int wave = threadIdx.x >> 5;                 // 0..3 (wave32)
  int z  = blockIdx.z;
  int bb = z / hdiv, hh = z % hdiv;
  const _Float16* Ap = A + bb * sAb + hh * sAh;
  const _Float16* Bp = Bt + bb * sBb + hh * sBh;
  long coff = bb * sCb + hh * sCh;

  int m0 = blockIdx.y * 64 + wave * 16;
  if (m0 >= Mm) return;                        // wave-uniform: EXEC stays all-ones
  int n0 = blockIdx.x * 64;
  int nt = (Nn - n0) >> 4; if (nt > 4) nt = 4; // wave-uniform

  v8f acc[4];
#pragma unroll
  for (int t = 0; t < 4; ++t)
#pragma unroll
    for (int i = 0; i < 8; ++i) acc[t][i] = 0.0f;

  // wave-uniform dispatch: each instantiation has unconditional WMMA updates
  if (nt >= 4)      gemm_kloop<4>(Ap, Bp, acc, m0, n0, Kk, lda, ldb, lane);
  else if (nt == 3) gemm_kloop<3>(Ap, Bp, acc, m0, n0, Kk, lda, ldb, lane);
  else if (nt == 2) gemm_kloop<2>(Ap, Bp, acc, m0, n0, Kk, lda, ldb, lane);
  else              gemm_kloop<1>(Ap, Bp, acc, m0, n0, Kk, lda, ldb, lane);

  int l = lane & 15, hi = lane >> 4;
#pragma unroll
  for (int t = 0; t < 4; ++t) {
    if (t >= nt) continue;                     // store-only guard (acc read-only here)
    int n = n0 + t * 16 + l;
    float bv = bias ? bias[n] : 0.0f;
#pragma unroll
    for (int r = 0; r < 8; ++r) {
      int m = m0 + hi * 8 + r;
      float v = acc[t][r] + bv;
      if (flags & 2) v = 0.5f * v * (1.0f + erff(v * 0.70710678118654752f)); // exact gelu
      long ci = coff + (long)m * ldc + n;
      if (Res) v += Res[ci];
      if (flags & 1) ((_Float16*)C)[ci] = (_Float16)v;
      else           ((float*)C)[ci]    = v;
    }
  }
}

// ---------------- weight convert f32(KxN) -> f16 N-major (NxK) ----------------
__global__ void wt_kernel(const float* __restrict__ W, _Float16* __restrict__ Wt, int K, int N) {
  size_t gid = (size_t)blockIdx.x * 256 + threadIdx.x;
  if (gid >= (size_t)K * N) return;
  int k = (int)(gid / N), n = (int)(gid % N);
  Wt[(size_t)n * K + k] = (_Float16)W[gid];
}

// ---------------- FPS (one block per frame) ----------------
__global__ void __launch_bounds__(256)
fps_kernel(const float* __restrict__ radar, int* __restrict__ aidx,
           float* __restrict__ anchors, float* __restrict__ xyzts)
{
  __shared__ float px[NPT], py[NPT], pz[NPT], dist[NPT];
  __shared__ float bd[256];
  __shared__ int   bi[256];
  __shared__ int   s_last;
  int f  = blockIdx.x;          // 0..7
  int b  = f / TQ, tq = f % TQ;
  int ct = 2 * tq;              // xyz time index (centers-1)
  int tid = threadIdx.x;
  const float* src = radar + (((size_t)b * TT + ct) * NPT) * 6;
  for (int n = tid; n < NPT; n += 256) {
    px[n] = src[n * 6 + 0]; py[n] = src[n * 6 + 1]; pz[n] = src[n * 6 + 2];
    dist[n] = 1e10f;
  }
  if (tid == 0) { s_last = 0; aidx[f * MA] = 0; }
  __syncthreads();
  for (int it = 1; it < MA; ++it) {
    int last = s_last;
    float lx = px[last], ly = py[last], lz = pz[last];
    float bestd = -1.0f; int besti = 0;
    for (int n = tid; n < NPT; n += 256) {
      float dx = px[n] - lx, dy = py[n] - ly, dz = pz[n] - lz;
      float d = fminf(dist[n], dx * dx + dy * dy + dz * dz);
      dist[n] = d;
      if (d > bestd) { bestd = d; besti = n; }   // strict > keeps first max
    }
    bd[tid] = bestd; bi[tid] = besti;
    __syncthreads();
    for (int s = 128; s > 0; s >>= 1) {
      if (tid < s) {
        if (bd[tid + s] > bd[tid] || (bd[tid + s] == bd[tid] && bi[tid + s] < bi[tid])) {
          bd[tid] = bd[tid + s]; bi[tid] = bi[tid + s];
        }
      }
      __syncthreads();
    }
    if (tid == 0) { s_last = bi[0]; aidx[f * MA + it] = bi[0]; }
    __syncthreads();
  }
  for (int m = tid; m < MA; m += 256) {
    int a = aidx[f * MA + m];
    float ax = px[a], ay = py[a], az = pz[a];
    anchors[((size_t)f * MA + m) * 3 + 0] = ax;
    anchors[((size_t)f * MA + m) * 3 + 1] = ay;
    anchors[((size_t)f * MA + m) * 3 + 2] = az;
    size_t xo = (((size_t)b * (TQ * MA)) + tq * MA + m) * 4;
    xyzts[xo + 0] = ax; xyzts[xo + 1] = ay; xyzts[xo + 2] = az;
    xyzts[xo + 3] = (float)(tq + 1);
  }
}

// ---------------- ball query: first K (by index) within radius ----------------
__global__ void ballq_kernel(const float* __restrict__ radar,
                             const float* __restrict__ anchors, int* __restrict__ nidx)
{
  int gid = blockIdx.x * 256 + threadIdx.x;
  if (gid >= BV * TQ * 3 * MA) return;
  int m  = gid % MA;
  int o  = (gid / MA) % 3;
  int tq = (gid / (MA * 3)) % TQ;
  int b  = gid / (MA * 3 * TQ);
  int j  = 2 * tq + o;                 // xyz_p index
  int jm = (j == 0) ? 0 : j - 1;       // xyz time index
  const float* pts = radar + (((size_t)b * TT + jm) * NPT) * 6;
  const float* an  = anchors + ((size_t)(b * TQ + tq) * MA + m) * 3;
  float ax = an[0], ay = an[1], az = an[2];
  int* idxp = nidx + (size_t)gid * KNB;
  int cnt = 0, first = 0;
  for (int n = 0; n < NPT; ++n) {
    float dx = pts[n * 6 + 0] - ax;
    float dy = pts[n * 6 + 1] - ay;
    float dz = pts[n * 6 + 2] - az;
    float d2 = dx * dx + dy * dy + dz * dz;
    if (d2 < RAD2) {
      if (cnt == 0) first = n;
      idxp[cnt++] = n;
      if (cnt == KNB) break;
    }
  }
  if (cnt == 0) { for (int k = 0; k < KNB; ++k) idxp[k] = 0; }
  else          { for (int k = cnt; k < KNB; ++k) idxp[k] = first; }
}

// ---------------- conv(7->1024) + max over 48 neighbors ----------------
__global__ void __launch_bounds__(256)
feat_kernel(const float* __restrict__ radar, const float* __restrict__ anchors,
            const int* __restrict__ nidx, const float* __restrict__ wd,
            const float* __restrict__ wf, float* __restrict__ feats)
{
  __shared__ float in7[48][8];
  int g  = blockIdx.x;               // 0..2047 : (b,tq,m)
  int m  = g % MA;
  int tq = (g / MA) % TQ;
  int b  = g / (MA * TQ);
  int tid = threadIdx.x;
  const float* an = anchors + ((size_t)(b * TQ + tq) * MA + m) * 3;
  float ax = an[0], ay = an[1], az = an[2];
  if (tid < 48) {
    int o = tid / KNB, k = tid % KNB;
    int j = 2 * tq + o, jm = (j == 0) ? 0 : j - 1;
    int n = nidx[((size_t)(((b * TQ + tq) * 3 + o) * MA + m)) * KNB + k];
    const float* p = radar + (((size_t)b * TT + jm) * NPT + n) * 6;
    in7[tid][0] = p[0] - ax;
    in7[tid][1] = p[1] - ay;
    in7[tid][2] = p[2] - az;
    in7[tid][3] = (float)(o - 1);
    in7[tid][4] = p[3];
    in7[tid][5] = p[4];
    in7[tid][6] = p[5];
  }
  __syncthreads();
  for (int d = tid; d < DIMM; d += 256) {
    float w0 = wd[d * 4 + 0], w1 = wd[d * 4 + 1], w2 = wd[d * 4 + 2], w3 = wd[d * 4 + 3];
    float w4 = wf[d * 3 + 0], w5 = wf[d * 3 + 1], w6 = wf[d * 3 + 2];
    float mx = -3.0e38f;
    for (int s = 0; s < 48; ++s) {
      float v = w0 * in7[s][0] + w1 * in7[s][1] + w2 * in7[s][2] + w3 * in7[s][3]
              + w4 * in7[s][4] + w5 * in7[s][5] + w6 * in7[s][6];
      mx = fmaxf(mx, v);
    }
    feats[(((size_t)b * (TQ * MA)) + tq * MA + m) * DIMM + d] = mx;
  }
}

// ---------------- embedding + joints -> padded x ----------------
__global__ void embed_kernel(const float* __restrict__ jt, const float* __restrict__ jp,
                             const float* __restrict__ pw, const float* __restrict__ pb,
                             const float* __restrict__ feats, const float* __restrict__ xyzts,
                             float* __restrict__ x)
{
  size_t gid = (size_t)blockIdx.x * 256 + threadIdx.x;
  if (gid >= (size_t)BV * SPAD * DIMM) return;
  int d = (int)(gid % DIMM);
  int s = (int)((gid / DIMM) % SPAD);
  int b = (int)(gid / ((size_t)DIMM * SPAD));
  float v;
  if (s < JNJ) {
    v = jt[s * DIMM + d] + jp[s * DIMM + d];
  } else if (s < SREAL) {
    int t = s - JNJ;
    const float* xr = xyzts + ((size_t)b * (TQ * MA) + t) * 4;
    v = xr[0] * pw[d * 4 + 0] + xr[1] * pw[d * 4 + 1] + xr[2] * pw[d * 4 + 2]
      + xr[3] * pw[d * 4 + 3] + pb[d] + feats[((size_t)b * (TQ * MA) + t) * DIMM + d];
  } else {
    v = 0.0f;                                  // pad rows
  }
  x[gid] = v;
}

// ---------------- LayerNorm (f32 row -> f16 row); pad rows -> 0 ----------------
__global__ void __launch_bounds__(256)
ln_f16_kernel(const float* __restrict__ x, const float* __restrict__ g,
              const float* __restrict__ bta, _Float16* __restrict__ out)
{
  __shared__ float sm[256];
  int row = blockIdx.x;
  int s = row % SPAD;
  int tid = threadIdx.x;
  _Float16* orow = out + (size_t)row * DIMM;
  if (s >= SREAL) {
    for (int d = tid; d < DIMM; d += 256) orow[d] = (_Float16)0.0f;
    return;
  }
  const float* xr = x + (size_t)row * DIMM;
  float v[4]; float sum = 0.0f;
#pragma unroll
  for (int i = 0; i < 4; ++i) { v[i] = xr[tid + i * 256]; sum += v[i]; }
  float mean = block_sum(sum, sm) * (1.0f / DIMM);
  float s2 = 0.0f;
#pragma unroll
  for (int i = 0; i < 4; ++i) { float d = v[i] - mean; s2 += d * d; }
  float var = block_sum(s2, sm) * (1.0f / DIMM);
  float r = rsqrtf(var + EPSV);
#pragma unroll
  for (int i = 0; i < 4; ++i) {
    int d = tid + i * 256;
    orow[d] = (_Float16)((v[i] - mean) * r * g[d] + bta[d]);
  }
}

// ---------------- transpose v -> (b,h,d,s) f16 ----------------
__global__ void vt_kernel(const _Float16* __restrict__ qkv, _Float16* __restrict__ vT) {
  size_t gid = (size_t)blockIdx.x * 256 + threadIdx.x;
  if (gid >= (size_t)BV * NHEADS * DHEADN * SPAD) return;
  int s = (int)(gid % SPAD);
  int d = (int)((gid / SPAD) % DHEADN);
  int h = (int)((gid / ((size_t)SPAD * DHEADN)) % NHEADS);
  int b = (int)(gid / ((size_t)SPAD * DHEADN * NHEADS));
  vT[gid] = qkv[((size_t)(b * SPAD + s)) * (3 * DIMM) + 2 * DIMM + h * DHEADN + d];
}

// --------- softmax (scale, mask j>=SREAL) + in-place f32 -> f16 probs ---------
__global__ void __launch_bounds__(128)
softmax_kernel(float* __restrict__ scores)
{
  __shared__ float sm[128];
  int i = blockIdx.x % SPAD;
  int z = blockIdx.x / SPAD;
  float* row = scores + ((size_t)z * SPAD + i) * SPAD;
  _Float16* prow = (_Float16*)row;    // row-local reuse; safe after in-reg loads
  int tid = threadIdx.x;
  if (i >= SREAL) {
    for (int j = tid; j < SPAD; j += 128) prow[j] = (_Float16)0.0f;
    return;
  }
  float v[9]; float mx = -3.0e38f;
#pragma unroll
  for (int t = 0; t < 9; ++t) {
    int j = tid + t * 128;
    if (j < SREAL) { v[t] = row[j] * ATT_SCALE; mx = fmaxf(mx, v[t]); }
    else v[t] = -3.0e38f;
  }
  mx = block_max(mx, sm);
  float e[9]; float sum = 0.0f;
#pragma unroll
  for (int t = 0; t < 9; ++t) {
    int j = tid + t * 128;
    e[t] = (j < SREAL) ? expf(v[t] - mx) : 0.0f;
    sum += e[t];
  }
  sum = block_sum(sum, sm);           // barriers => all reads done before writes
  float inv = 1.0f / sum;
#pragma unroll
  for (int t = 0; t < 9; ++t) {
    int j = tid + t * 128;
    if (j < SPAD) prow[j] = (_Float16)(e[t] * inv);
  }
}

// ---------------- head MLPs (one block per (b,joint)) ----------------
__global__ void __launch_bounds__(256)
head_kernel(const float* __restrict__ x,
            const float* rls, const float* rlb, const float* rw1, const float* rb1,
            const float* rw2, const float* rb2, const float* hls, const float* hlb,
            const float* hw1, const float* hb1, const float* hw2, const float* hb2,
            float* __restrict__ j64_out, float* __restrict__ pose_out)
{
  __shared__ float sm[256];
  __shared__ float hn[DIMM];
  __shared__ float t1[512];
  __shared__ float j64s[64];
  __shared__ float mv[2];
  __shared__ float t2[32];
  int g = blockIdx.x;
  int b = g / JNJ, jt = g % JNJ;
  int tid = threadIdx.x;
  const float* xr = x + ((size_t)b * SPAD + jt) * DIMM;
  float v[4]; float sum = 0.0f;
#pragma unroll
  for (int i = 0; i < 4; ++i) { v[i] = xr[tid + i * 256]; sum += v[i]; }
  float mean = block_sum(sum, sm) * (1.0f / DIMM);
  float s2 = 0.0f;
#pragma unroll
  for (int i = 0; i < 4; ++i) { float d = v[i] - mean; s2 += d * d; }
  float var = block_sum(s2, sm) * (1.0f / DIMM);
  float r = rsqrtf(var + EPSV);
#pragma unroll
  for (int i = 0; i < 4; ++i) {
    int d = tid + i * 256;
    hn[d] = (v[i] - mean) * r * rls[d] + rlb[d];
  }
  __syncthreads();
  for (int o = tid; o < 512; o += 256) {
    float acc = rb1[o];
    for (int i = 0; i < DIMM; ++i) acc += hn[i] * rw1[i * 512 + o];
    t1[o] = fmaxf(acc, 0.0f);
  }
  __syncthreads();
  if (tid < 64) {
    float acc = rb2[tid];
    for (int i = 0; i < 512; ++i) acc += t1[i] * rw2[i * 64 + tid];
    j64s[tid] = acc;
    j64_out[(size_t)g * 64 + tid] = acc;
  }
  __syncthreads();
  if (tid == 0) {
    float s = 0.0f;
    for (int i = 0; i < 64; ++i) s += j64s[i];
    float m = s * (1.0f / 64.0f);
    float q = 0.0f;
    for (int i = 0; i < 64; ++i) { float d = j64s[i] - m; q += d * d; }
    mv[0] = m; mv[1] = rsqrtf(q * (1.0f / 64.0f) + EPSV);
  }
  __syncthreads();
  if (tid < 32) {
    float acc = hb1[tid];
    for (int i = 0; i < 64; ++i)
      acc += ((j64s[i] - mv[0]) * mv[1] * hls[i] + hlb[i]) * hw1[i * 32 + tid];
    t2[tid] = fmaxf(acc, 0.0f);
  }
  __syncthreads();
  if (tid < 3) {
    float acc = hb2[tid];
    for (int i = 0; i < 32; ++i) acc += t2[i] * hw2[i * 3 + tid];
    pose_out[(size_t)g * 3 + tid] = acc;
  }
}

// ================= host launcher =================
extern "C" void kernel_launch(void* const* d_in, const int* in_sizes, int n_in,
                              void* d_out, int out_size, void* d_ws, size_t ws_size,
                              hipStream_t stream) {
  (void)in_sizes; (void)n_in; (void)out_size; (void)ws_size;
  const float* radar    = (const float*)d_in[0];
  const float* conv_d_w = (const float*)d_in[1];
  const float* conv_f_w = (const float*)d_in[2];
  const float* pos_w    = (const float*)d_in[3];
  const float* pos_b    = (const float*)d_in[4];
  const float* joint_t  = (const float*)d_in[5];
  const float* joint_p  = (const float*)d_in[6];
  const float* ln1_s    = (const float*)d_in[7];
  const float* ln1_b    = (const float*)d_in[8];
  const float* qkv_w    = (const float*)d_in[9];
  const float* out_w    = (const float*)d_in[10];
  const float* out_b    = (const float*)d_in[11];
  const float* ln2_s    = (const float*)d_in[12];
  const float* ln2_b    = (const float*)d_in[13];
  const float* ff_w1    = (const float*)d_in[14];
  const float* ff_b1    = (const float*)d_in[15];
  const float* ff_w2    = (const float*)d_in[16];
  const float* ff_b2    = (const float*)d_in[17];
  const float* red_ln_s = (const float*)d_in[18];
  const float* red_ln_b = (const float*)d_in[19];
  const float* red_w1   = (const float*)d_in[20];
  const float* red_b1   = (const float*)d_in[21];
  const float* red_w2   = (const float*)d_in[22];
  const float* red_b2   = (const float*)d_in[23];
  const float* head_ln_s= (const float*)d_in[24];
  const float* head_ln_b= (const float*)d_in[25];
  const float* head_w1  = (const float*)d_in[26];
  const float* head_b1  = (const float*)d_in[27];
  const float* head_w2  = (const float*)d_in[28];
  const float* head_b2  = (const float*)d_in[29];

  float* outp = (float*)d_out;
  float* out_pose  = outp;                       // (2,17,3)   = 102
  float* out_j64   = outp + 102;                 // (2,17,64)  = 2176
  float* out_feat  = outp + 2278;                // (2,1024,1024)
  float* out_xyzts = outp + 2278 + (size_t)BV * TQ * MA * DIMM;  // (2,1024,4)

  // ---- workspace carve ----
  char* wsp = (char*)d_ws;
  auto carve = [&](size_t bytes) -> char* {
    char* p = wsp; wsp += (bytes + 255) & ~(size_t)255; return p;
  };
  float*    x      = (float*)   carve((size_t)BV * SPAD * DIMM * 4);
  _Float16* h16    = (_Float16*)carve((size_t)BV * SPAD * DIMM * 2);
  _Float16* qkv16  = (_Float16*)carve((size_t)BV * SPAD * 3 * DIMM * 2);
  _Float16* vT     = (_Float16*)carve((size_t)BV * NHEADS * DHEADN * SPAD * 2);
  float*    scores = (float*)   carve((size_t)BV * NHEADS * SPAD * SPAD * 4);
  _Float16* o16    = (_Float16*)carve((size_t)BV * SPAD * DIMM * 2);
  _Float16* t16    = (_Float16*)carve((size_t)BV * SPAD * MLPN * 2);
  _Float16* wt16   = (_Float16*)carve((size_t)5 * 8388608 * 2);
  float*    anchors= (float*)   carve((size_t)BV * TQ * MA * 3 * 4);
  int*      aidx   = (int*)     carve((size_t)BV * TQ * MA * 4);
  int*      nidx   = (int*)     carve((size_t)BV * TQ * 3 * MA * KNB * 4);

  const size_t WL = 8388608;           // halfs per layer block
  const size_t OFF_OUT = 3145728, OFF_FF1 = 4194304, OFF_FF2 = 6291456;

  // ---- weight conversion / transpose (f32 KxN -> f16 NxK) ----
  for (int l = 0; l < 5; ++l) {
    _Float16* wl = wt16 + (size_t)l * WL;
    wt_kernel<<<(1024 * 3072 + 255) / 256, 256, 0, stream>>>(qkv_w + (size_t)l * 1024 * 3072, wl, 1024, 3072);
    wt_kernel<<<(1024 * 1024 + 255) / 256, 256, 0, stream>>>(out_w + (size_t)l * 1024 * 1024, wl + OFF_OUT, 1024, 1024);
    wt_kernel<<<(1024 * 2048 + 255) / 256, 256, 0, stream>>>(ff_w1 + (size_t)l * 1024 * 2048, wl + OFF_FF1, 1024, 2048);
    wt_kernel<<<(2048 * 1024 + 255) / 256, 256, 0, stream>>>(ff_w2 + (size_t)l * 2048 * 1024, wl + OFF_FF2, 2048, 1024);
  }

  // ---- preprocessing ----
  fps_kernel  <<<BV * TQ, 256, 0, stream>>>(radar, aidx, anchors, out_xyzts);
  ballq_kernel<<<(BV * TQ * 3 * MA + 255) / 256, 256, 0, stream>>>(radar, anchors, nidx);
  feat_kernel <<<BV * TQ * MA, 256, 0, stream>>>(radar, anchors, nidx, conv_d_w, conv_f_w, out_feat);
  embed_kernel<<<(int)(((size_t)BV * SPAD * DIMM + 255) / 256), 256, 0, stream>>>(
      joint_t, joint_p, pos_w, pos_b, out_feat, out_xyzts, x);

  const int MR = BV * SPAD;            // 2112 rows (multiple of 64)
  // ---- transformer ----
  for (int l = 0; l < 5; ++l) {
    _Float16* wl = wt16 + (size_t)l * WL;
    // LN1 -> h16
    ln_f16_kernel<<<MR, 256, 0, stream>>>(x, ln1_s + l * DIMM, ln1_b + l * DIMM, h16);
    // qkv = h16 @ Wqkv  (f16 out)
    gemm_wmma_kernel<<<dim3(3 * DIMM / 64, MR / 64, 1), 128, 0, stream>>>(
        h16, wl, nullptr, nullptr, qkv16, MR, 3 * DIMM, DIMM, DIMM, DIMM, 3 * DIMM,
        1, 0, 0, 0, 0, 0, 0, 1);
    // v transpose
    vt_kernel<<<(int)(((size_t)BV * NHEADS * DHEADN * SPAD + 255) / 256), 256, 0, stream>>>(qkv16, vT);
    // scores[z] = q @ k^T  (batched over 16 (b,h))
    gemm_wmma_kernel<<<dim3((SPAD + 63) / 64, (SPAD + 63) / 64, BV * NHEADS), 128, 0, stream>>>(
        qkv16, qkv16 + DIMM, nullptr, nullptr, scores,
        SPAD, SPAD, DHEADN, 3 * DIMM, 3 * DIMM, SPAD, NHEADS,
        (long)SPAD * 3 * DIMM, (long)DHEADN,
        (long)SPAD * 3 * DIMM, (long)DHEADN,
        (long)NHEADS * SPAD * SPAD, (long)SPAD * SPAD, 0);
    // softmax + f16 probs (row-local in-place)
    softmax_kernel<<<BV * NHEADS * SPAD, 128, 0, stream>>>(scores);
    // o[z] = probs @ v  (f16 out into (b,s,h*128+d))
    gemm_wmma_kernel<<<dim3(DHEADN / 64, (SPAD + 63) / 64, BV * NHEADS), 128, 0, stream>>>(
        (const _Float16*)scores, vT, nullptr, nullptr, o16,
        SPAD, DHEADN, SPAD, 2 * SPAD, SPAD, DIMM, NHEADS,
        2L * NHEADS * SPAD * SPAD, 2L * SPAD * SPAD,
        (long)NHEADS * DHEADN * SPAD, (long)DHEADN * SPAD,
        (long)SPAD * DIMM, (long)DHEADN, 1);
    // x += o @ Wout + b
    gemm_wmma_kernel<<<dim3(DIMM / 64, MR / 64, 1), 128, 0, stream>>>(
        o16, wl + OFF_OUT, out_b + l * DIMM, x, x, MR, DIMM, DIMM, DIMM, DIMM, DIMM,
        1, 0, 0, 0, 0, 0, 0, 0);
    // LN2 -> h16
    ln_f16_kernel<<<MR, 256, 0, stream>>>(x, ln2_s + l * DIMM, ln2_b + l * DIMM, h16);
    // t = gelu(h16 @ Wff1 + b1) (f16 out)
    gemm_wmma_kernel<<<dim3(MLPN / 64, MR / 64, 1), 128, 0, stream>>>(
        h16, wl + OFF_FF1, ff_b1 + l * MLPN, nullptr, t16, MR, MLPN, DIMM, DIMM, DIMM, MLPN,
        1, 0, 0, 0, 0, 0, 0, 3);
    // x += t @ Wff2 + b2
    gemm_wmma_kernel<<<dim3(DIMM / 64, MR / 64, 1), 128, 0, stream>>>(
        t16, wl + OFF_FF2, ff_b2 + l * DIMM, x, x, MR, DIMM, MLPN, MLPN, MLPN, DIMM,
        1, 0, 0, 0, 0, 0, 0, 0);
  }

  // ---- heads ----
  head_kernel<<<BV * JNJ, 256, 0, stream>>>(
      x, red_ln_s, red_ln_b, red_w1, red_b1, red_w2, red_b2,
      head_ln_s, head_ln_b, head_w1, head_b1, head_w2, head_b2,
      out_j64, out_pose);
}
